// AdaGCL_42932493091129
// MI455X (gfx1250) — matrix operand order repaired and checked by reading, not compiled, not run on hardware
//
#include <hip/hip_runtime.h>
#include <hip/hip_bf16.h>
#include <math.h>

#define N_NODES 150000
#define N_EDGES 1000000
#define DIM 64

typedef __attribute__((ext_vector_type(16))) _Float16 v16h;
typedef __attribute__((ext_vector_type(8)))  float    v8f;

// ---------------------------------------------------------------------------
// Convert the four 64x64 f32 weight matrices into WMMA-ready f16 fragments.
// Fragment layout (flat half index):
//   ((((layer*2 + w)*2 + ks)*4 + nt)*32 + lane)*16 + i
// where B-fragment element i for lane holds W[ks*32 + (lane<16?0:8) + (i<8?i:i+8)][nt*16 + (lane&15)]
// (B matrix 32x16, 16-bit: lanes 0-15 hold K=0..7,16..23, lanes 16-31 hold K=8..15,24..31,
//  column = lane % 16 — matches the A fragment K order.)
// ---------------------------------------------------------------------------
__global__ void convert_weights_kernel(const float* __restrict__ nbW0,
                                       const float* __restrict__ selfW0,
                                       const float* __restrict__ nbW1,
                                       const float* __restrict__ selfW1,
                                       _Float16* __restrict__ frag) {
    int tid = blockIdx.x * blockDim.x + threadIdx.x;
    if (tid >= 16384) return;
    int i     = tid & 15;
    int lane  = (tid >> 4)  & 31;
    int nt    = (tid >> 9)  & 3;
    int ks    = (tid >> 11) & 1;
    int w     = (tid >> 12) & 1;
    int layer = (tid >> 13) & 1;
    const float* W = layer ? (w ? selfW1 : nbW1) : (w ? selfW0 : nbW0);
    int k = ks * 32 + ((lane < 16) ? 0 : 8) + ((i < 8) ? i : i + 8);
    int n = nt * 16 + (lane & 15);
    frag[tid] = (_Float16)W[k * 64 + n];
}

// ---------------------------------------------------------------------------
// Per-node attention scalars via WMMA:
//   a1[n] = relu(x[n] @ nbW   + nbb  ) . attW[0:64]
//   a2[n] = relu(x[n] @ selfW + selfb) . attW[64:128]
// One wave (32 threads) per 16-node tile; 16 v_wmma_f32_16x16x32_f16 per wave.
// ---------------------------------------------------------------------------
#define PUT4(v, base, f4)                       \
    v[(base) + 0] = (_Float16)(f4).x;           \
    v[(base) + 1] = (_Float16)(f4).y;           \
    v[(base) + 2] = (_Float16)(f4).z;           \
    v[(base) + 3] = (_Float16)(f4).w;

__global__ void ascore_kernel(const float* __restrict__ X,
                              const _Float16* __restrict__ frag,  // this layer: 8192 halfs
                              const float* __restrict__ nbb,
                              const float* __restrict__ selfb,
                              const float* __restrict__ attW,     // 128 floats
                              float* __restrict__ a1,
                              float* __restrict__ a2) {
    __shared__ float T[16 * 64];
    const int tile = blockIdx.x;                 // 9375 tiles, exact
    const int lane = threadIdx.x & 31;
    const int row  = tile * 16 + (lane & 15);

    // ---- A fragments (ISA 16-bit A 16x32 layout) ----
    // lanes 0-15: elems 0..7 = K 0..7,  elems 8..15 = K 16..23  (row = lane)
    // lanes16-31: elems 0..7 = K 8..15, elems 8..15 = K 24..31  (row = lane-16)
    const float4* xr = (const float4*)(X + (size_t)row * 64);
    const int cb = (lane < 16) ? 0 : 2;          // float4-chunk base (K offset /4)
    float4 c0  = xr[cb + 0],  c1  = xr[cb + 1];
    float4 c4  = xr[cb + 4],  c5  = xr[cb + 5];
    float4 c8  = xr[cb + 8],  c9  = xr[cb + 9];
    float4 c12 = xr[cb + 12], c13 = xr[cb + 13];
    v16h a_lo, a_hi;
    PUT4(a_lo, 0, c0)  PUT4(a_lo, 4, c1)  PUT4(a_lo, 8, c4)   PUT4(a_lo, 12, c5)
    PUT4(a_hi, 0, c8)  PUT4(a_hi, 4, c9)  PUT4(a_hi, 8, c12)  PUT4(a_hi, 12, c13)

    for (int w = 0; w < 2; ++w) {
        const _Float16* fw = frag + w * 4096;
#pragma unroll
        for (int nt = 0; nt < 4; ++nt) {
            v16h b0 = *(const v16h*)(fw +        (nt * 32 + lane) * 16);  // K 0..31
            v16h b1 = *(const v16h*)(fw + 2048 + (nt * 32 + lane) * 16);  // K 32..63
            v8f c = {};
            c = __builtin_amdgcn_wmma_f32_16x16x32_f16(false, a_lo, false, b0,
                                                       (short)0, c, false, false);
            c = __builtin_amdgcn_wmma_f32_16x16x32_f16(false, a_hi, false, b1,
                                                       (short)0, c, false, false);
            // C/D layout: VGPR j -> M = j (+8 for lanes 16-31), N = lane & 15
            const int mbase = (lane < 16) ? 0 : 8;
            const int ncol  = nt * 16 + (lane & 15);
#pragma unroll
            for (int j = 0; j < 8; ++j) T[(mbase + j) * 64 + ncol] = c[j];
        }
        __syncthreads();
        if (lane < 16) {
            const float* bias = w ? selfb : nbb;
            const float* aw   = attW + w * 64;
            float acc = 0.f;
#pragma unroll 8
            for (int n = 0; n < 64; ++n) {
                float h = T[lane * 64 + n] + bias[n];
                h = h > 0.f ? h : 0.f;
                acc += h * aw[n];
            }
            (w ? a2 : a1)[row] = acc;
        }
        __syncthreads();
    }
}

// ---------------------------------------------------------------------------
// Per-edge hard-concrete gate + degree accumulation.
// ---------------------------------------------------------------------------
__global__ void mask_kernel(const int* __restrict__ row, const int* __restrict__ col,
                            const float* __restrict__ noise,
                            const float* __restrict__ a1, const float* __restrict__ a2,
                            const float* __restrict__ attb,
                            float* __restrict__ maskb, float* __restrict__ deg) {
    int e = blockIdx.x * blockDim.x + threadIdx.x;
    if (e >= N_EDGES) return;
    float u = noise[e];
    u = fminf(fmaxf(u, 1e-7f), 1.0f - 1e-7f);
    float logit_u = logf(u) - log1pf(-u);
    int r = row[e], c = col[e];
    float z    = logit_u + a1[r] + a2[c] + attb[0];
    float gate = 1.0f / (1.0f + expf(-z));
    float m    = gate * 1.5f - 0.45f;              // * (ZETA-GAMMA) + GAMMA
    m = fminf(fmaxf(m, 0.0f), 1.0f);
    maskb[e] = m;
    atomicAdd(&deg[r], m);
}

// deg[n] -> clip((deg[n] + EPS_DEG)^-0.5, 0, 10), in place
__global__ void dinv_kernel(float* __restrict__ deg) {
    int n = blockIdx.x * blockDim.x + threadIdx.x;
    if (n >= N_NODES) return;
    float rs = deg[n] + 1e-6f;
    float d  = 1.0f / sqrtf(rs);
    deg[n] = fminf(d, 10.0f);
}

// ---------------------------------------------------------------------------
// SpMM scatter: out[row] += mask_e * dinv[row] * dinv[col] * x[col].
// 16 threads per edge, 4 dims each (float4 gather, 4 f32 atomics).
// ---------------------------------------------------------------------------
__global__ void spmm_kernel(const int* __restrict__ row, const int* __restrict__ col,
                            const float* __restrict__ maskb,
                            const float* __restrict__ dinv,
                            const float* __restrict__ X, float* __restrict__ out) {
    int tid = blockIdx.x * blockDim.x + threadIdx.x;   // E*16 threads, exact grid
    int e   = tid >> 4;
    int sub = tid & 15;
    if (e >= N_EDGES) return;
    int r = row[e], c = col[e];
    float val = maskb[e] * dinv[r] * dinv[c];
    float4 xv = *(const float4*)(X + (size_t)c * 64 + sub * 4);
    float* o  = out + (size_t)r * 64 + sub * 4;
    atomicAdd(o + 0, val * xv.x);
    atomicAdd(o + 1, val * xv.y);
    atomicAdd(o + 2, val * xv.z);
    atomicAdd(o + 3, val * xv.w);
}

// out = x0 + x1 (elementwise over N*64)
__global__ void add_kernel(const float* __restrict__ x0, const float* __restrict__ x1,
                           float* __restrict__ out) {
    int i = blockIdx.x * blockDim.x + threadIdx.x;     // 9.6M, exact grid
    out[i] = x0[i] + x1[i];
}

// ---------------------------------------------------------------------------
extern "C" void kernel_launch(void* const* d_in, const int* in_sizes, int n_in,
                              void* d_out, int out_size, void* d_ws, size_t ws_size,
                              hipStream_t stream) {
    const float* X0     = (const float*)d_in[0];
    const int*   row    = (const int*)d_in[1];
    const int*   col    = (const int*)d_in[2];
    const float* noise0 = (const float*)d_in[3];
    const float* noise1 = (const float*)d_in[4];
    const float* nbW0   = (const float*)d_in[5];
    const float* nbb0   = (const float*)d_in[6];
    const float* selfW0 = (const float*)d_in[7];
    const float* selfb0 = (const float*)d_in[8];
    const float* attW0  = (const float*)d_in[9];
    const float* attb0  = (const float*)d_in[10];
    const float* nbW1   = (const float*)d_in[11];
    const float* nbb1   = (const float*)d_in[12];
    const float* selfW1 = (const float*)d_in[13];
    const float* selfb1 = (const float*)d_in[14];
    const float* attW1  = (const float*)d_in[15];
    const float* attb1  = (const float*)d_in[16];
    float* out = (float*)d_out;

    // Workspace layout (frag first so v16h loads stay 32B-aligned):
    char* w = (char*)d_ws;
    _Float16* frag  = (_Float16*)w;  w += 16384 * sizeof(_Float16);        // 32 KB
    float*    x1    = (float*)w;     w += (size_t)N_NODES * DIM * 4;       // 38.4 MB
    float*    maskb = (float*)w;     w += (size_t)N_EDGES * 4;             // 4 MB
    float*    a1    = (float*)w;     w += (size_t)N_NODES * 4;
    float*    a2    = (float*)w;     w += (size_t)N_NODES * 4;
    float*    deg   = (float*)w;     w += (size_t)N_NODES * 4;

    hipMemsetAsync(x1,  0, (size_t)N_NODES * DIM * 4, stream);
    hipMemsetAsync(deg, 0, (size_t)N_NODES * 4, stream);

    convert_weights_kernel<<<64, 256, 0, stream>>>(nbW0, selfW0, nbW1, selfW1, frag);

    // ---- layer 1 (input x0) ----
    ascore_kernel<<<N_NODES / 16, 32, 0, stream>>>(X0, frag, nbb0, selfb0, attW0, a1, a2);
    mask_kernel<<<(N_EDGES + 255) / 256, 256, 0, stream>>>(row, col, noise0, a1, a2,
                                                           attb0, maskb, deg);
    dinv_kernel<<<(N_NODES + 255) / 256, 256, 0, stream>>>(deg);
    spmm_kernel<<<(N_EDGES * 16) / 256, 256, 0, stream>>>(row, col, maskb, deg, X0, x1);

    // out = x0 + x1 (layer-2 scatter then accumulates x2 on top)
    add_kernel<<<(N_NODES * DIM) / 256, 256, 0, stream>>>(X0, x1, out);

    // ---- layer 2 (input x1) ----
    hipMemsetAsync(deg, 0, (size_t)N_NODES * 4, stream);
    ascore_kernel<<<N_NODES / 16, 32, 0, stream>>>(x1, frag + 8192, nbb1, selfb1, attW1, a1, a2);
    mask_kernel<<<(N_EDGES + 255) / 256, 256, 0, stream>>>(row, col, noise1, a1, a2,
                                                           attb1, maskb, deg);
    dinv_kernel<<<(N_NODES + 255) / 256, 256, 0, stream>>>(deg);
    spmm_kernel<<<(N_EDGES * 16) / 256, 256, 0, stream>>>(row, col, maskb, deg, x1, out);
}
